// CrossAxisAttention_13469017440620
// MI455X (gfx1250) — matrix operand order repaired and checked by reading, not compiled, not run on hardware
//
#include <hip/hip_runtime.h>

typedef __attribute__((ext_vector_type(16))) __bf16 v16bf;
typedef __attribute__((ext_vector_type(8)))  __bf16 v8bf;
typedef __attribute__((ext_vector_type(8)))  float  v8f;

#define DEV __device__ __forceinline__

DEV __bf16 f2bf(float f) {
    unsigned u = __builtin_bit_cast(unsigned, f);
    unsigned r = u + 0x7FFFu + ((u >> 16) & 1u);
    unsigned short h = (unsigned short)(r >> 16);
    return __builtin_bit_cast(__bf16, h);
}
DEV __bf16 to_bf(float f)  { return f2bf(f); }
DEV __bf16 to_bf(__bf16 x) { return x; }

DEV v8f wmma_bf16(v16bf a, v16bf b, v8f c) {
    return __builtin_amdgcn_wmma_f32_16x16x32_bf16(false, a, false, b, (short)0, c, false, false);
}

DEV v16bf cat8(v8bf lo, v8bf hi) {
    return __builtin_shufflevector(lo, hi, 0,1,2,3,4,5,6,7,8,9,10,11,12,13,14,15);
}

// A-operand fragment (16x32 bf16, M x K): lane m = lane&15;
// elems e -> k = half*16 + e (16 contiguous)
DEV v16bf fragA(const __bf16* base, int ld) {
    int lane = threadIdx.x & 31;
    const __bf16* p = base + (lane & 15) * ld + ((lane >> 4) << 4);
    v8bf lo = *(const v8bf*)p;
    v8bf hi = *(const v8bf*)(p + 8);
    return cat8(lo, hi);
}

// B-operand fragment (32x16 bf16 as [n][k] row-major): lane n = lane&15;
// elems e -> k = half*16 + e (16 contiguous)
DEV v16bf fragB(const __bf16* base, int ld) {
    int lane = threadIdx.x & 31;
    const __bf16* p = base + (lane & 15) * ld + ((lane >> 4) << 4);
    v8bf lo = *(const v8bf*)p;
    v8bf hi = *(const v8bf*)(p + 8);
    return cat8(lo, hi);
}

DEV v8f zero8() {
    v8f z;
#pragma unroll
    for (int i = 0; i < 8; ++i) z[i] = 0.f;
    return z;
}

// ---- CDNA5 async global->LDS copy (ASYNCcnt) -------------------------------
// dst = workgroup-relative LDS byte address (flat shared ptr low 32 bits),
// src = 64-bit global address; per-lane addressed, 16 bytes.
DEV void async_b128(unsigned dst, const void* src) {
    asm volatile("global_load_async_to_lds_b128 %0, %1, off"
                 :: "v"(dst), "v"(src) : "memory");
}
DEV unsigned ldsoff(const void* p) { return (unsigned)(unsigned long long)p; }
DEV void wait_async0() { asm volatile("s_wait_asynccnt 0x0" ::: "memory"); }
DEV void wait_async4() { asm volatile("s_wait_asynccnt 0x4" ::: "memory"); }
DEV void wait_async8() { asm volatile("s_wait_asynccnt 0x8" ::: "memory"); }

// ---------------------------------------------------------------------------
// Generic bf16 WMMA GEMM:  C(MxN) = A(MxK, fp32) * B   (+bias, +addsrc)
//   TRANSB=1: Bsrc is (K x N) row-major (e.g. the input image (C,N))
//   TRANSB=0: Bsrc is (N x K) row-major (already B-operand layout)
// Output: exactly one of outF (fp32 MxN), outNM (bf16 NxM), outMN (bf16 MxN).
// Block 256 thr (8 waves), tile 64(M) x 64(N), waves 2x4, 2 MMA tiles each.
// ---------------------------------------------------------------------------
template<typename TB, bool TRANSB>
__global__ __launch_bounds__(256) void gemm_wmma(
    const float* __restrict__ A, const TB* __restrict__ Bsrc,
    float* __restrict__ outF, __bf16* __restrict__ outNM, __bf16* __restrict__ outMN,
    const float* __restrict__ bias, const float* __restrict__ addsrc,
    int M, int N, int K)
{
    constexpr int LDT = 40;
    __shared__ __attribute__((aligned(16))) __bf16 sA[64 * LDT];
    __shared__ __attribute__((aligned(16))) __bf16 sB[64 * LDT];
    int tid  = threadIdx.x;
    int wave = tid >> 5, lane = tid & 31;
    int m0 = blockIdx.x * 64, n0 = blockIdx.y * 64;
    int wm = (wave >> 2) * 32, wn = (wave & 3) * 16;
    v8f c0 = zero8(), c1 = zero8();

    for (int k0 = 0; k0 < K; k0 += 32) {
        {   // A tile 64x32, row-major
            int r = tid >> 2, cb = (tid & 3) * 8;
            const float* g = A + (size_t)(m0 + r) * K + k0 + cb;
            __bf16* dst = sA + r * LDT + cb;
#pragma unroll
            for (int i = 0; i < 8; ++i) dst[i] = f2bf(g[i]);
        }
        if (TRANSB) {   // sB[n][k] = Bsrc[k0+k][n0+n]
            int kk = tid >> 3, nb = (tid & 7) * 8;
            const TB* g = Bsrc + (size_t)(k0 + kk) * N + n0 + nb;
#pragma unroll
            for (int i = 0; i < 8; ++i) sB[(nb + i) * LDT + kk] = to_bf(g[i]);
        } else {        // sB[n][k] = Bsrc[n0+n][k0+k]
            int r = tid >> 2, cb = (tid & 3) * 8;
            const TB* g = Bsrc + (size_t)(n0 + r) * K + k0 + cb;
            __bf16* dst = sB + r * LDT + cb;
#pragma unroll
            for (int i = 0; i < 8; ++i) dst[i] = to_bf(g[i]);
        }
        __syncthreads();
        v16bf bf = fragB(sB + wn * LDT, LDT);
        v16bf a0 = fragA(sA + wm * LDT, LDT);
        v16bf a1 = fragA(sA + (wm + 16) * LDT, LDT);
        c0 = wmma_bf16(a0, bf, c0);
        c1 = wmma_bf16(a1, bf, c1);
        __syncthreads();
    }

    int hh = lane >> 4;
    int nOut = n0 + wn + (lane & 15);
#pragma unroll
    for (int t = 0; t < 2; ++t) {
        int mBase = m0 + wm + t * 16;
        v8f c = t ? c1 : c0;
#pragma unroll
        for (int v = 0; v < 8; ++v) {
            int mOut = mBase + hh * 8 + v;
            float val = c[v];
            if (bias)   val += bias[mOut];
            if (addsrc) val += addsrc[(size_t)mOut * N + nOut];
            if (outF)        outF[(size_t)mOut * N + nOut] = val;
            else if (outNM)  outNM[(size_t)nOut * M + mOut] = f2bf(val);
            else             outMN[(size_t)mOut * N + nOut] = f2bf(val);
        }
    }
}

DEV int diag_idx(int p) { return (p & 63) - (p >> 6) + 63; }   // in [0,126]

// ---------------------------------------------------------------------------
// Flash attention, d=128. Q,K: (N x 128) bf16 ; V: (128 x N) bf16 ; O: (N x 128) f32.
// Computes S^T = K*Q^T (lane owns one query), online softmax, O^T = V*P^T.
// Double-buffered LDS staging via GLOBAL_LOAD_ASYNC_TO_LDS_B128 (ASYNCcnt):
// prefetch chunk j+1 while computing chunk j. gather!=0 routes K rows / V cols
// through diag_idx (V falls back to synchronous scalar staging).
// Block = 128 thr (4 waves x 16 queries), grid.x covers query tiles of 64.
// ---------------------------------------------------------------------------
__global__ __launch_bounds__(128) void flash_attn(
    const __bf16* __restrict__ Q, const __bf16* __restrict__ K,
    const __bf16* __restrict__ V, float* __restrict__ O,
    const float* __restrict__ wdir, int widx,
    int N, int gather, int qlimit, int accum)
{
    constexpr int LDK = 136, LDV = 40;
    constexpr int KB = 32 * LDK, VB = 128 * LDV;
    __shared__ __attribute__((aligned(16))) __bf16 sK[2 * KB];
    __shared__ __attribute__((aligned(16))) __bf16 sV[2 * VB];
    int tid = threadIdx.x, wave = tid >> 5, lane = tid & 31, hl = lane >> 4;
    int q0 = blockIdx.x * 64 + wave * 16;

    v16bf qf[4];
#pragma unroll
    for (int ch = 0; ch < 4; ++ch)
        qf[ch] = fragB(Q + (size_t)q0 * 128 + ch * 32, 128);

    v8f o[8];
#pragma unroll
    for (int t = 0; t < 8; ++t) o[t] = zero8();
    float Mq = -1e30f, Lq = 0.f;
    const float scl = 0.08838834764831845f;   // 1/sqrt(128)

    int kr = tid >> 2, kseg = (tid & 3) * 32;

    auto stage = [&](int buf, int j0) {
        {   // 32 K rows x 128 d : 4 x b128 per thread, async
            int src = gather ? diag_idx(j0 + kr) : (j0 + kr);
            const __bf16* g = K + (size_t)src * 128 + kseg;
            unsigned dst = ldsoff(sK + buf * KB + kr * LDK + kseg);
#pragma unroll
            for (int i = 0; i < 4; ++i) async_b128(dst + i * 16, g + i * 8);
        }
        if (!gather) {   // V[128][32] : 4 x b128 per thread, async
            const __bf16* g = V + (size_t)tid * N + j0;
            unsigned dst = ldsoff(sV + buf * VB + tid * LDV);
#pragma unroll
            for (int i = 0; i < 4; ++i) async_b128(dst + i * 16, g + i * 8);
        } else {         // gathered V columns: scalar DS staging
            __bf16* dstp = sV + buf * VB + tid * LDV;
            const __bf16* g = V + (size_t)tid * N;
            for (int j = 0; j < 32; ++j) dstp[j] = g[diag_idx(j0 + j)];
        }
    };

    stage(0, 0);
    for (int j0 = 0; j0 < N; j0 += 32) {
        int buf = (j0 >> 5) & 1;
        int nxt = j0 + 32;
        if (nxt < N) {
            stage(buf ^ 1, nxt);               // prefetch next chunk
            if (gather) wait_async4(); else wait_async8();   // current chunk landed
        } else {
            wait_async0();
        }
        __syncthreads();
        const __bf16* bK = sK + buf * KB;
        const __bf16* bV = sV + buf * VB;

        // hoist all 8 K fragments, then 8 back-to-back WMMAs
        v16bf ka[8];
#pragma unroll
        for (int ch = 0; ch < 4; ++ch) {
            ka[2 * ch]     = fragA(bK + ch * 32, LDK);
            ka[2 * ch + 1] = fragA(bK + 16 * LDK + ch * 32, LDK);
        }
        v8f s0 = zero8(), s1 = zero8();
#pragma unroll
        for (int ch = 0; ch < 4; ++ch) {
            s0 = wmma_bf16(ka[2 * ch],     qf[ch], s0);
            s1 = wmma_bf16(ka[2 * ch + 1], qf[ch], s1);
        }

        float mloc = -1e30f;
#pragma unroll
        for (int v = 0; v < 8; ++v) {
            s0[v] *= scl; s1[v] *= scl;
            mloc = fmaxf(mloc, fmaxf(s0[v], s1[v]));
        }
        mloc = fmaxf(mloc, __shfl_xor(mloc, 16));
        float mnew = fmaxf(Mq, mloc);
        float corr = __expf(Mq - mnew);
        float p0[8], p1[8], ssum = 0.f;
#pragma unroll
        for (int v = 0; v < 8; ++v) {
            p0[v] = __expf(s0[v] - mnew);
            p1[v] = __expf(s1[v] - mnew);
            ssum += p0[v] + p1[v];
        }
        ssum += __shfl_xor(ssum, 16);
        Lq = Lq * corr + ssum;
        Mq = mnew;
#pragma unroll
        for (int t = 0; t < 8; ++t)
#pragma unroll
            for (int v = 0; v < 8; ++v) o[t][v] *= corr;

        // P^T C-fragment -> B-fragment (keys 0..31) via half-exchange
        v16bf pf;
#pragma unroll
        for (int v = 0; v < 8; ++v) {
            float t0x = __shfl_xor(p0[v], 16);
            float t1x = __shfl_xor(p1[v], 16);
            pf[v]     = f2bf(hl ? t1x   : p0[v]);
            pf[v + 8] = f2bf(hl ? p1[v] : t0x);
        }
        // hoist all 8 V fragments, then 8 back-to-back WMMAs
        v16bf va[8];
#pragma unroll
        for (int t = 0; t < 8; ++t) va[t] = fragA(bV + t * 16 * LDV, LDV);
#pragma unroll
        for (int t = 0; t < 8; ++t) o[t] = wmma_bf16(va[t], pf, o[t]);
        __syncthreads();
    }

    int qn = q0 + (lane & 15);
    if (qn < qlimit) {
        float inv = wdir[widx] / Lq;
#pragma unroll
        for (int t = 0; t < 8; ++t)
#pragma unroll
            for (int v = 0; v < 8; ++v) {
                size_t idx = (size_t)qn * 128 + t * 16 + hl * 8 + v;
                float val = o[t][v] * inv;
                if (accum) O[idx] += val; else O[idx] = val;
            }
    }
}

// ---------------------------------------------------------------------------
// Axial attention: one 64-long sequence per block (4 waves x 16 queries).
// rowmode=1: positions s*64+c ; rowmode=0: positions c*64+s. Accumulates O.
// ---------------------------------------------------------------------------
__global__ __launch_bounds__(128) void axial_attn(
    const __bf16* __restrict__ Q, const __bf16* __restrict__ K,
    const __bf16* __restrict__ V, float* __restrict__ O,
    const float* __restrict__ wdir, int widx, int rowmode)
{
    constexpr int LDK = 136, LDV = 72, NPOS = 4096;
    __shared__ __attribute__((aligned(16))) __bf16 sK[64 * LDK];
    __shared__ __attribute__((aligned(16))) __bf16 sV[128 * LDV];
    int tid = threadIdx.x, wave = tid >> 5, lane = tid & 31, hl = lane >> 4;
    int s = blockIdx.x;
    auto pos = [&](int c) { return rowmode ? s * 64 + c : c * 64 + s; };

    {   // stage K: 64 rows x 128
        int r = tid >> 1, seg = (tid & 1) * 64;
        const __bf16* g = K + (size_t)pos(r) * 128 + seg;
        __bf16* dst = sK + r * LDK + seg;
#pragma unroll
        for (int i = 0; i < 8; ++i)
            *(v8bf*)(dst + i * 8) = *(const v8bf*)(g + i * 8);
    }
    {   // stage V: 128 rows x 64 seq cols
        __bf16* dst = sV + tid * LDV;
        const __bf16* g = V + (size_t)tid * NPOS;
        if (rowmode) {
#pragma unroll
            for (int i = 0; i < 8; ++i)
                *(v8bf*)(dst + i * 8) = *(const v8bf*)(g + s * 64 + i * 8);
        } else {
            for (int j = 0; j < 64; ++j) dst[j] = g[j * 64 + s];
        }
    }
    int qn = wave * 16 + (lane & 15);
    v16bf qf[4];
#pragma unroll
    for (int ch = 0; ch < 4; ++ch) {
        const __bf16* p = Q + (size_t)pos(qn) * 128 + ch * 32 + hl * 16;
        qf[ch] = cat8(*(const v8bf*)p, *(const v8bf*)(p + 8));
    }
    __syncthreads();

    const float scl = 0.08838834764831845f;
    v8f sc[4];
#pragma unroll
    for (int kt = 0; kt < 4; ++kt) {
        sc[kt] = zero8();
#pragma unroll
        for (int ch = 0; ch < 4; ++ch) {
            v16bf a = fragA(sK + kt * 16 * LDK + ch * 32, LDK);
            sc[kt] = wmma_bf16(a, qf[ch], sc[kt]);
        }
    }
    float mloc = -1e30f;
#pragma unroll
    for (int kt = 0; kt < 4; ++kt)
#pragma unroll
        for (int v = 0; v < 8; ++v) {
            sc[kt][v] *= scl;
            mloc = fmaxf(mloc, sc[kt][v]);
        }
    mloc = fmaxf(mloc, __shfl_xor(mloc, 16));
    float p[4][8], ssum = 0.f;
#pragma unroll
    for (int kt = 0; kt < 4; ++kt)
#pragma unroll
        for (int v = 0; v < 8; ++v) {
            p[kt][v] = __expf(sc[kt][v] - mloc);
            ssum += p[kt][v];
        }
    ssum += __shfl_xor(ssum, 16);

    v8f o[8];
#pragma unroll
    for (int t = 0; t < 8; ++t) o[t] = zero8();
#pragma unroll
    for (int jc = 0; jc < 2; ++jc) {
        v16bf pf;
#pragma unroll
        for (int v = 0; v < 8; ++v) {
            float t0x = __shfl_xor(p[2 * jc][v], 16);
            float t1x = __shfl_xor(p[2 * jc + 1][v], 16);
            pf[v]     = f2bf(hl ? t1x : p[2 * jc][v]);
            pf[v + 8] = f2bf(hl ? p[2 * jc + 1][v] : t0x);
        }
#pragma unroll
        for (int t = 0; t < 8; ++t) {
            v16bf a = fragA(sV + t * 16 * LDV + jc * 32, LDV);
            o[t] = wmma_bf16(a, pf, o[t]);
        }
    }
    float inv = wdir[widx] / ssum;
    size_t base = (size_t)pos(qn) * 128;
#pragma unroll
    for (int t = 0; t < 8; ++t)
#pragma unroll
        for (int v = 0; v < 8; ++v)
            O[base + t * 16 + hl * 8 + v] += o[t][v] * inv;
}

__global__ void softmax4(const float* __restrict__ a, float* __restrict__ w) {
    if (threadIdx.x == 0 && blockIdx.x == 0) {
        float m = fmaxf(fmaxf(a[0], a[1]), fmaxf(a[2], a[3]));
        float e0 = __expf(a[0] - m), e1 = __expf(a[1] - m);
        float e2 = __expf(a[2] - m), e3 = __expf(a[3] - m);
        float inv = 1.f / (e0 + e1 + e2 + e3);
        w[0] = e0 * inv; w[1] = e1 * inv; w[2] = e2 * inv; w[3] = e3 * inv;
    }
}

extern "C" void kernel_launch(void* const* d_in, const int* in_sizes, int n_in,
                              void* d_out, int out_size, void* d_ws, size_t ws_size,
                              hipStream_t stream) {
    (void)in_sizes; (void)n_in; (void)out_size; (void)ws_size;
    const int B = 2, NPOS = 4096, D = 128, CL = 64, CH = 192;

    const float* ll  = (const float*)d_in[0];
    const float* hf  = (const float*)d_in[1];
    const float* Wqh = (const float*)d_in[2];
    const float* Wkh = (const float*)d_in[3];
    const float* Wvh = (const float*)d_in[4];
    const float* Wql = (const float*)d_in[5];
    const float* Wkl = (const float*)d_in[6];
    const float* Wvl = (const float*)d_in[7];
    const float* Wph = (const float*)d_in[8];
    const float* bph = (const float*)d_in[9];
    const float* Wpl = (const float*)d_in[10];
    const float* bpl = (const float*)d_in[11];
    const float* Wah = (const float*)d_in[12];
    const float* bah = (const float*)d_in[13];
    const float* Wal = (const float*)d_in[14];
    const float* bal = (const float*)d_in[15];
    const float* alpha = (const float*)d_in[16];

    float* ll_out = (float*)d_out;                       // (B, CL, N)
    float* hf_out = ll_out + (size_t)B * CL * NPOS;      // (B, CH, N)

    char* ws = (char*)d_ws;
    auto alloc = [&](size_t bytes) {
        void* p = (void*)ws;
        ws += (bytes + 255) & ~(size_t)255;
        return p;
    };
    size_t qe = (size_t)B * NPOS * D;
    float*  wdir = (float*)alloc(16);
    __bf16* Qh = (__bf16*)alloc(qe * 2);
    __bf16* Kh = (__bf16*)alloc(qe * 2);
    __bf16* Vh = (__bf16*)alloc(qe * 2);
    __bf16* Ql = (__bf16*)alloc(qe * 2);
    __bf16* Kl = (__bf16*)alloc(qe * 2);
    __bf16* Vl = (__bf16*)alloc(qe * 2);
    float*  Ol = (float*)alloc(qe * 4);
    float*  Oh = (float*)alloc(qe * 4);
    __bf16* tl = (__bf16*)alloc((size_t)B * NPOS * CL * 2);
    __bf16* th = (__bf16*)alloc((size_t)B * NPOS * CH * 2);

    softmax4<<<dim3(1), dim3(32), 0, stream>>>(alpha, wdir);

    dim3 blk(256);
    for (int b = 0; b < B; ++b) {
        const float* xh = hf + (size_t)b * CH * NPOS;
        const float* xl = ll + (size_t)b * CL * NPOS;
        __bf16 *Qhb = Qh + (size_t)b * NPOS * D, *Khb = Kh + (size_t)b * NPOS * D;
        __bf16 *Vhb = Vh + (size_t)b * NPOS * D, *Qlb = Ql + (size_t)b * NPOS * D;
        __bf16 *Klb = Kl + (size_t)b * NPOS * D, *Vlb = Vl + (size_t)b * NPOS * D;
        float  *Olb = Ol + (size_t)b * NPOS * D, *Ohb = Oh + (size_t)b * NPOS * D;
        __bf16 *tlb = tl + (size_t)b * NPOS * CL, *thb = th + (size_t)b * NPOS * CH;

        // ---- projections: Q/K -> (N x d) bf16, V -> (d x N) bf16 ----
        gemm_wmma<float, true><<<dim3(2, 64), blk, 0, stream>>>(Wqh, xh, nullptr, Qhb, nullptr, nullptr, nullptr, D, NPOS, CH);
        gemm_wmma<float, true><<<dim3(2, 64), blk, 0, stream>>>(Wkh, xh, nullptr, Khb, nullptr, nullptr, nullptr, D, NPOS, CH);
        gemm_wmma<float, true><<<dim3(2, 64), blk, 0, stream>>>(Wvh, xh, nullptr, nullptr, Vhb, nullptr, nullptr, D, NPOS, CH);
        gemm_wmma<float, true><<<dim3(2, 64), blk, 0, stream>>>(Wql, xl, nullptr, Qlb, nullptr, nullptr, nullptr, D, NPOS, CL);
        gemm_wmma<float, true><<<dim3(2, 64), blk, 0, stream>>>(Wkl, xl, nullptr, Klb, nullptr, nullptr, nullptr, D, NPOS, CL);
        gemm_wmma<float, true><<<dim3(2, 64), blk, 0, stream>>>(Wvl, xl, nullptr, nullptr, Vlb, nullptr, nullptr, D, NPOS, CL);

        // ---- global attention (initializes O with wdir[3]*result) ----
        flash_attn<<<dim3(64), dim3(128), 0, stream>>>(Qhb, Klb, Vlb, Olb, wdir, 3, NPOS, 0, NPOS, 0);
        flash_attn<<<dim3(64), dim3(128), 0, stream>>>(Qlb, Khb, Vhb, Ohb, wdir, 3, NPOS, 0, NPOS, 0);

        // ---- axial row (widx 0) and column (widx 1), accumulate ----
        axial_attn<<<dim3(64), dim3(128), 0, stream>>>(Qhb, Klb, Vlb, Olb, wdir, 0, 1);
        axial_attn<<<dim3(64), dim3(128), 0, stream>>>(Qlb, Khb, Vhb, Ohb, wdir, 0, 1);
        axial_attn<<<dim3(64), dim3(128), 0, stream>>>(Qhb, Klb, Vlb, Olb, wdir, 1, 0);
        axial_attn<<<dim3(64), dim3(128), 0, stream>>>(Qlb, Khb, Vhb, Ohb, wdir, 1, 0);

        // ---- diag variant: 127 distinct queries, gathered K/V, accumulate ----
        flash_attn<<<dim3(2), dim3(128), 0, stream>>>(Qhb, Klb, Vlb, Olb, wdir, 2, NPOS, 1, 127, 1);
        flash_attn<<<dim3(2), dim3(128), 0, stream>>>(Qlb, Khb, Vhb, Ohb, wdir, 2, NPOS, 1, 127, 1);

        // ---- t = Wp*O^T + bp + x   -> (N x C) bf16 ----
        gemm_wmma<float, false><<<dim3(1, 64), blk, 0, stream>>>(Wpl, Olb, nullptr, tlb, nullptr, bpl, xl, CL, NPOS, D);
        gemm_wmma<float, false><<<dim3(3, 64), blk, 0, stream>>>(Wph, Ohb, nullptr, thb, nullptr, bph, xh, CH, NPOS, D);

        // ---- y = Wa*t + ba  -> fp32 output ----
        gemm_wmma<__bf16, false><<<dim3(1, 64), blk, 0, stream>>>(Wal, tlb, ll_out + (size_t)b * CL * NPOS, nullptr, nullptr, bal, nullptr, CL, NPOS, CL);
        gemm_wmma<__bf16, false><<<dim3(3, 64), blk, 0, stream>>>(Wah, thb, hf_out + (size_t)b * CH * NPOS, nullptr, nullptr, bah, nullptr, CH, NPOS, CH);
    }
}
